// AdaptiveHierarchicalQuantizer_13262859010400
// MI455X (gfx1250) — matrix operand order repaired and checked by reading, not compiled, not run on hardware
//
#include <hip/hip_runtime.h>
#include <hip/hip_bf16.h>

typedef __bf16 bf16_t;
typedef __attribute__((ext_vector_type(16))) __bf16 v16bf;
typedef __attribute__((ext_vector_type(8)))  __bf16 v8bf;
typedef __attribute__((ext_vector_type(8)))  float  v8f;

#define N_TOK   32768
#define EDIM    512
#define NE      1024
#define NUMQ    3
#define BETA    0.25f
#define SKEPS   0.05f
#define SKITERS 10

// ---- float <-> order-preserving signed int key (for atomic min/max on floats) ----
__device__ __forceinline__ int f2k(float f) {
    int i = __float_as_int(f);
    return i < 0 ? (i ^ 0x7fffffff) : i;
}
__device__ __forceinline__ float k2f(int k) {
    return __int_as_float(k < 0 ? (k ^ 0x7fffffff) : k);
}

// =====================================================================
// init: zero x_q accumulator (lives in d_out), loss accumulator, colsum
// =====================================================================
__global__ __launch_bounds__(256) void k_zero(float* __restrict__ xq,
                                              float* __restrict__ lossacc,
                                              float* __restrict__ colsum) {
    size_t gid = (size_t)blockIdx.x * 256 + threadIdx.x;
    xq[gid] = 0.0f;
    if (gid < NE) colsum[gid] = 0.0f;
    if (gid == 0) *lossacc = 0.0f;
}

// convert all codebooks fp32 -> bf16 once
__global__ __launch_bounds__(256) void k_cvt_cb(const float* __restrict__ cb,
                                                bf16_t* __restrict__ cbh) {
    size_t gid = (size_t)blockIdx.x * 256 + threadIdx.x;
    cbh[gid] = (bf16_t)cb[gid];
}

// per-layer: cn[c] = ||cb_c||^2
__global__ __launch_bounds__(256) void k_cn(const float* __restrict__ cb,
                                            float* __restrict__ cn) {
    int c = blockIdx.x * 256 + threadIdx.x;
    const float* row = cb + (size_t)c * EDIM;
    float s = 0.0f;
    for (int k = 0; k < EDIM; ++k) { float t = row[k]; s += t * t; }
    cn[c] = s;
}

// per-layer: xn[r] = ||x_r - xq_r||^2   (one wave per row)
__global__ __launch_bounds__(256) void k_xn(const float* __restrict__ x,
                                            const float* __restrict__ xq,
                                            float* __restrict__ xn) {
    int lane = threadIdx.x & 31, wave = threadIdx.x >> 5;
    int r = blockIdx.x * 8 + wave;
    size_t base = (size_t)r * EDIM;
    float s = 0.0f;
    for (int e = lane; e < EDIM; e += 32) {
        float d = x[base + e] - xq[base + e];
        s += d * d;
    }
    for (int off = 16; off; off >>= 1) s += __shfl_xor(s, off, 32);
    if (lane == 0) xn[r] = s;
}

// per-layer: v = 1, min/max keys reset
__global__ __launch_bounds__(256) void k_layer_init(float* __restrict__ v,
                                                    int* __restrict__ maxk,
                                                    int* __restrict__ mink) {
    for (int c = threadIdx.x; c < NE; c += 256) v[c] = 1.0f;
    if (threadIdx.x == 0) {
        *maxk = f2k(-3.4e38f);
        *mink = f2k(3.4e38f);
    }
}

// =====================================================================
// WMMA distance GEMM: d[r,c] = xn[r] + cn[c] - 2 * (x-xq)_r . cb_c
// 64x64 tile / block (4 waves), K staged in LDS as bf16, fused min/max.
// =====================================================================
__global__ __launch_bounds__(128)
void k_gemm(const float* __restrict__ x, const float* __restrict__ xq,
            const bf16_t* __restrict__ cbh, const float* __restrict__ xn,
            const float* __restrict__ cn, float* __restrict__ Kmat,
            int* __restrict__ maxk, int* __restrict__ mink) {
    __shared__ __align__(16) bf16_t Ab[64][32];   // [row][k]
    __shared__ __align__(16) bf16_t Bb[64][32];   // [col][k]
    __shared__ int smax[4], smin[4];

    const int tid  = threadIdx.x;
    const int lane = tid & 31;
    const int wave = tid >> 5;
    const int rowBase = blockIdx.y * 64;
    const int colBase = blockIdx.x * 64;

    v8f acc0 = {}, acc1 = {}, acc2 = {}, acc3 = {};

    const int ldrow = tid >> 1;          // 0..63
    const int ldk   = (tid & 1) * 16;    // 0 or 16

    for (int kk = 0; kk < EDIM; kk += 32) {
        // ---- stage A = (x - xq) as bf16 ----
        {
            const float4* px = (const float4*)(x  + (size_t)(rowBase + ldrow) * EDIM + kk + ldk);
            const float4* pq = (const float4*)(xq + (size_t)(rowBase + ldrow) * EDIM + kk + ldk);
            float rv[16];
#pragma unroll
            for (int j = 0; j < 4; ++j) {
                float4 a = px[j], b = pq[j];
                rv[4 * j + 0] = a.x - b.x; rv[4 * j + 1] = a.y - b.y;
                rv[4 * j + 2] = a.z - b.z; rv[4 * j + 3] = a.w - b.w;
            }
            v8bf lo, hi;
#pragma unroll
            for (int j = 0; j < 8; ++j) { lo[j] = (bf16_t)rv[j]; hi[j] = (bf16_t)rv[8 + j]; }
            *(v8bf*)&Ab[ldrow][ldk]     = lo;
            *(v8bf*)&Ab[ldrow][ldk + 8] = hi;
        }
        // ---- stage B = cbh tile ----
        {
            const v8bf* pb = (const v8bf*)(cbh + (size_t)(colBase + ldrow) * EDIM + kk + ldk);
            *(v8bf*)&Bb[ldrow][ldk]     = pb[0];
            *(v8bf*)&Bb[ldrow][ldk + 8] = pb[1];
        }
        __syncthreads();
        {
            const int m  = (wave << 4) | (lane & 15);
            const int h  = lane >> 4;
            const int nb = lane & 15;
            union F { v16bf v; v8bf p[2]; };
            F a;
            a.p[0] = *(const v8bf*)&Ab[m][16 * h];
            a.p[1] = *(const v8bf*)&Ab[m][16 * h + 8];
            F b0, b1, b2, b3;
            b0.p[0] = *(const v8bf*)&Bb[nb][16 * h];      b0.p[1] = *(const v8bf*)&Bb[nb][16 * h + 8];
            b1.p[0] = *(const v8bf*)&Bb[16 + nb][16 * h]; b1.p[1] = *(const v8bf*)&Bb[16 + nb][16 * h + 8];
            b2.p[0] = *(const v8bf*)&Bb[32 + nb][16 * h]; b2.p[1] = *(const v8bf*)&Bb[32 + nb][16 * h + 8];
            b3.p[0] = *(const v8bf*)&Bb[48 + nb][16 * h]; b3.p[1] = *(const v8bf*)&Bb[48 + nb][16 * h + 8];
            acc0 = __builtin_amdgcn_wmma_f32_16x16x32_bf16(false, a.v, false, b0.v, (short)0, acc0, false, false);
            acc1 = __builtin_amdgcn_wmma_f32_16x16x32_bf16(false, a.v, false, b1.v, (short)0, acc1, false, false);
            acc2 = __builtin_amdgcn_wmma_f32_16x16x32_bf16(false, a.v, false, b2.v, (short)0, acc2, false, false);
            acc3 = __builtin_amdgcn_wmma_f32_16x16x32_bf16(false, a.v, false, b3.v, (short)0, acc3, false, false);
        }
        __syncthreads();
    }

    // ---- epilogue: d = xn + cn - 2*acc, write + fused min/max ----
    const int h  = lane >> 4;
    const int nb = lane & 15;
    float xnv[8];
#pragma unroll
    for (int p = 0; p < 8; ++p)
        xnv[p] = xn[rowBase + (wave << 4) + (h << 3) + p];

    float lmax = -3.4e38f, lmin = 3.4e38f;
    v8f accs[4] = {acc0, acc1, acc2, acc3};
#pragma unroll
    for (int nt = 0; nt < 4; ++nt) {
        int gn = colBase + (nt << 4) + nb;
        float cnv = cn[gn];
#pragma unroll
        for (int p = 0; p < 8; ++p) {
            int gm = rowBase + (wave << 4) + (h << 3) + p;
            float dv = xnv[p] + cnv - 2.0f * accs[nt][p];
            Kmat[(size_t)gm * NE + gn] = dv;
            lmax = fmaxf(lmax, dv);
            lmin = fminf(lmin, dv);
        }
    }
    for (int off = 16; off; off >>= 1) {
        lmax = fmaxf(lmax, __shfl_xor(lmax, off, 32));
        lmin = fminf(lmin, __shfl_xor(lmin, off, 32));
    }
    if (lane == 0) { smax[wave] = f2k(lmax); smin[wave] = f2k(lmin); }
    __syncthreads();
    if (tid == 0) {
        int mx = max(max(smax[0], smax[1]), max(smax[2], smax[3]));
        int mn = min(min(smin[0], smin[1]), min(smin[2], smin[3]));
        atomicMax(maxk, mx);
        atomicMin(mink, mn);
    }
}

// center + exponentiate in place: K = exp(-((d-mid)/amp)/eps)
__global__ __launch_bounds__(256) void k_expc(float* __restrict__ Kmat,
                                              const int* __restrict__ maxk,
                                              const int* __restrict__ mink) {
    float mx = k2f(*maxk), mn = k2f(*mink);
    float mid = 0.5f * (mx + mn);
    float amp = mx - mid + 1e-5f;
    float sc  = 1.0f / (amp * SKEPS);
    size_t gid = (size_t)blockIdx.x * 256 + threadIdx.x;
    Kmat[gid] = __expf(-(Kmat[gid] - mid) * sc);
}

// Sinkhorn row step: u[r] = 1 / sum_c K[r,c] v[c]   (one wave per row)
__global__ __launch_bounds__(256) void k_rowKv(const float* __restrict__ Kmat,
                                               const float* __restrict__ v,
                                               float* __restrict__ u) {
    __shared__ float vs[NE];
    for (int i = threadIdx.x; i < NE; i += 256) vs[i] = v[i];
    __syncthreads();
    int lane = threadIdx.x & 31, wave = threadIdx.x >> 5;
    int r = blockIdx.x * 8 + wave;
    const float* row = Kmat + (size_t)r * NE;
    float s = 0.0f;
    for (int c = lane; c < NE; c += 32) s += row[c] * vs[c];
    for (int off = 16; off; off >>= 1) s += __shfl_xor(s, off, 32);
    if (lane == 0) u[r] = 1.0f / s;
}

// Sinkhorn col step (partial): colsum[c] += sum_{r in stripe} K[r,c] u[r]
__global__ __launch_bounds__(256) void k_colKu(const float* __restrict__ Kmat,
                                               const float* __restrict__ u,
                                               float* __restrict__ colsum) {
    __shared__ float us[512];
    int c  = blockIdx.x * 256 + threadIdx.x;
    int r0 = blockIdx.y * 512;
    for (int i = threadIdx.x; i < 512; i += 256) us[i] = u[r0 + i];
    __syncthreads();
    float s = 0.0f;
    for (int j = 0; j < 512; ++j)
        s += Kmat[(size_t)(r0 + j) * NE + c] * us[j];
    atomicAdd(&colsum[c], s);
}

// v = 1/colsum, and re-zero colsum for next iteration
__global__ __launch_bounds__(256) void k_vinv(float* __restrict__ colsum,
                                              float* __restrict__ v) {
    int c = blockIdx.x * 256 + threadIdx.x;
    float s = colsum[c];
    v[c] = 1.0f / s;
    colsum[c] = 0.0f;
}

// argmax_c v[c]*K[r,c]  (Q = diag(u) K diag(v); u constant per row)
__global__ __launch_bounds__(256) void k_argmax(const float* __restrict__ Kmat,
                                                const float* __restrict__ v,
                                                int* __restrict__ idx,
                                                float* __restrict__ outIdx,
                                                int layer) {
    __shared__ float vs[NE];
    for (int i = threadIdx.x; i < NE; i += 256) vs[i] = v[i];
    __syncthreads();
    int lane = threadIdx.x & 31, wave = threadIdx.x >> 5;
    int r = blockIdx.x * 8 + wave;
    const float* row = Kmat + (size_t)r * NE;
    float bv = -3.4e38f; int bi = 0;
    for (int c = lane; c < NE; c += 32) {
        float q = row[c] * vs[c];
        if (q > bv) { bv = q; bi = c; }
    }
    for (int off = 16; off; off >>= 1) {
        float ov = __shfl_xor(bv, off, 32);
        int   oi = __shfl_xor(bi, off, 32);
        if (ov > bv || (ov == bv && oi < bi)) { bv = ov; bi = oi; }
    }
    if (lane == 0) {
        idx[r] = bi;
        outIdx[(size_t)r * NUMQ + layer] = (float)bi;
    }
}

// gather + residual loss + accumulate x_q
__global__ __launch_bounds__(256) void k_update(const float* __restrict__ x,
                                                const float* __restrict__ cb,
                                                const int* __restrict__ idx,
                                                float* __restrict__ xq,
                                                float* __restrict__ lossacc) {
    __shared__ float red[8];
    size_t gid = (size_t)blockIdx.x * 256 + threadIdx.x;
    int r = (int)(gid >> 9);     // / EDIM
    int e = (int)(gid & 511);
    int c = idx[r];
    float q   = cb[(size_t)c * EDIM + e];
    float xo  = xq[gid];
    float diff = q - (x[gid] - xo);   // x_q - residual
    xq[gid] = xo + q;
    float s = diff * diff;
    int lane = threadIdx.x & 31, wave = threadIdx.x >> 5;
    for (int off = 16; off; off >>= 1) s += __shfl_xor(s, off, 32);
    if (lane == 0) red[wave] = s;
    __syncthreads();
    if (threadIdx.x == 0) {
        float t = 0.0f;
        for (int i = 0; i < 8; ++i) t += red[i];
        atomicAdd(lossacc, t);
    }
}

__global__ void k_finalize(const float* __restrict__ lossacc,
                           float* __restrict__ lossOut) {
    // each layer loss = (1+BETA)*mean(diff^2); mean over NUMQ layers
    *lossOut = (*lossacc) * (1.0f + BETA) / ((float)NUMQ * (float)N_TOK * (float)EDIM);
}

// =====================================================================
extern "C" void kernel_launch(void* const* d_in, const int* in_sizes, int n_in,
                              void* d_out, int out_size, void* d_ws, size_t ws_size,
                              hipStream_t stream) {
    (void)in_sizes; (void)n_in; (void)out_size; (void)ws_size;
    const float* x   = (const float*)d_in[0];
    const float* cbf = (const float*)d_in[1];

    // ---- workspace layout ----
    char* w = (char*)d_ws;
    float* Kmat = (float*)w;  w += (size_t)N_TOK * NE * 4;        // 128 MB
    bf16_t* cbh = (bf16_t*)w; w += (size_t)NUMQ * NE * EDIM * 2;  // 3 MB
    float* xn = (float*)w;    w += (size_t)N_TOK * 4;
    float* cn = (float*)w;    w += (size_t)NE * 4;
    float* u  = (float*)w;    w += (size_t)N_TOK * 4;
    float* v  = (float*)w;    w += (size_t)NE * 4;
    float* colsum = (float*)w; w += (size_t)NE * 4;
    int*   idx = (int*)w;     w += (size_t)N_TOK * 4;
    int*   maxk = (int*)w;    w += 64;
    int*   mink = (int*)w;    w += 64;
    float* lossacc = (float*)w; w += 64;

    // ---- output layout: [x_q | loss | indices] ----
    float* xq      = (float*)d_out;
    float* lossOut = xq + (size_t)N_TOK * EDIM;
    float* idxOut  = lossOut + 1;

    const size_t NXE = (size_t)N_TOK * EDIM;   // 16,777,216

    k_zero<<<(int)(NXE / 256), 256, 0, stream>>>(xq, lossacc, colsum);
    k_cvt_cb<<<(int)((size_t)NUMQ * NE * EDIM / 256), 256, 0, stream>>>(cbf, cbh);

    for (int layer = 0; layer < NUMQ; ++layer) {
        const float*  cbl  = cbf + (size_t)layer * NE * EDIM;
        const bf16_t* cbhl = cbh + (size_t)layer * NE * EDIM;

        k_cn<<<NE / 256, 256, 0, stream>>>(cbl, cn);
        k_xn<<<N_TOK / 8, 256, 0, stream>>>(x, xq, xn);
        k_layer_init<<<1, 256, 0, stream>>>(v, maxk, mink);

        dim3 gg(NE / 64, N_TOK / 64);   // (16, 512)
        k_gemm<<<gg, 128, 0, stream>>>(x, xq, cbhl, xn, cn, Kmat, maxk, mink);

        k_expc<<<(int)((size_t)N_TOK * NE / 256), 256, 0, stream>>>(Kmat, maxk, mink);

        for (int it = 0; it < SKITERS; ++it) {
            k_rowKv<<<N_TOK / 8, 256, 0, stream>>>(Kmat, v, u);
            k_colKu<<<dim3(NE / 256, N_TOK / 512), 256, 0, stream>>>(Kmat, u, colsum);
            k_vinv<<<NE / 256, 256, 0, stream>>>(colsum, v);
        }

        k_argmax<<<N_TOK / 8, 256, 0, stream>>>(Kmat, v, idx, idxOut, layer);
        k_update<<<(int)(NXE / 256), 256, 0, stream>>>(x, cbl, idx, xq, lossacc);
    }

    k_finalize<<<1, 1, 0, stream>>>(lossacc, lossOut);
}
